// MultiHeadedSelfAttention_79276506350218
// MI455X (gfx1250) — matrix operand order repaired and checked by reading, not compiled, not run on hardware
//
#include <hip/hip_runtime.h>

#define B_  4
#define S_  2048
#define D_  768
#define H_  12
#define DH_ 64

typedef __attribute__((ext_vector_type(16))) _Float16 v16h;
typedef __attribute__((ext_vector_type(8)))  _Float16 h8;
typedef __attribute__((ext_vector_type(8)))  float    v8f;
typedef __attribute__((ext_vector_type(4)))  unsigned u32x4;
typedef __attribute__((ext_vector_type(4)))  int      i32x4;
typedef __attribute__((ext_vector_type(8)))  int      i32x8;

#if defined(__has_builtin)
#  if __has_builtin(__builtin_amdgcn_tensor_load_to_lds)
#    define USE_TDM 1
#  endif
#endif
#ifndef USE_TDM
#  define USE_TDM 0
#endif

static __device__ __forceinline__ v16h pack16(h8 lo, h8 hi) {
  v16h r;
#pragma unroll
  for (int i = 0; i < 8; ++i) { r[i] = lo[i]; r[8 + i] = hi[i]; }
  return r;
}

#if USE_TDM
// Issue one 2D TDM tile load: tile (tile_d0 x tile_d1) elements of 2 bytes from
// a row-major tensor with row stride stride0, into LDS at lds_addr (packed rows).
// D# bitfields per CDNA5 ISA.
static __device__ __forceinline__ void tdm_load_2d(unsigned lds_addr,
                                                   const void* gptr,
                                                   unsigned tensor_d0, unsigned tensor_d1,
                                                   unsigned tile_d0, unsigned tile_d1,
                                                   unsigned stride0)
{
  const unsigned long long ga = (unsigned long long)(size_t)gptr;
  u32x4 g0;
  g0[0] = 1u;                                                // count=1, user D#
  g0[1] = lds_addr;                                          // lds_addr[31:0]
  g0[2] = (unsigned)(ga & 0xffffffffu);                      // global_addr[31:0]
  g0[3] = (unsigned)((ga >> 32) & 0x01ffffffu) | (2u << 30); // global_addr[56:32] | type=2
  i32x8 g1;
  g1[0] = (int)(1u << 16);                                   // data_size=1 (2 bytes)
  g1[1] = (int)((tensor_d0 & 0xffffu) << 16);                // tensor_dim0[15:0]
  g1[2] = (int)(((tensor_d0 >> 16) & 0xffffu) | ((tensor_d1 & 0xffffu) << 16));
  g1[3] = (int)(((tensor_d1 >> 16) & 0xffffu) | ((tile_d0 & 0xffffu) << 16));
  g1[4] = (int)(tile_d1 & 0xffffu);                          // tile_dim1 | tile_dim2=0
  g1[5] = (int)stride0;                                      // tensor_dim0_stride[31:0]
  g1[6] = 0;
  g1[7] = 0;
  const i32x4 z4 = {0, 0, 0, 0};
#if __clang_major__ >= 23
  const i32x8 z8 = {0, 0, 0, 0, 0, 0, 0, 0};
  __builtin_amdgcn_tensor_load_to_lds(g0, g1, z4, z4, z8, 0);
#else
  __builtin_amdgcn_tensor_load_to_lds(g0, g1, z4, z4, 0);
#endif
}
#endif

// ---------------------------------------------------------------------------
// Stage 0: one-shot f32 -> f16 conversion (x and the three weight matrices),
// so the GEMM inner loop is pure b128 loads + WMMA with no per-tile cvt.
// ---------------------------------------------------------------------------
__global__ __launch_bounds__(256)
void cvt_f16(const float* __restrict__ src, _Float16* __restrict__ dst, int n)
{
  const int i = (blockIdx.x * 256 + threadIdx.x) * 8;
  if (i + 8 <= n) {
    const float4 a = *(const float4*)(src + i);
    const float4 b = *(const float4*)(src + i + 4);
    h8 o;
    o[0] = (_Float16)a.x; o[1] = (_Float16)a.y; o[2] = (_Float16)a.z; o[3] = (_Float16)a.w;
    o[4] = (_Float16)b.x; o[5] = (_Float16)b.y; o[6] = (_Float16)b.z; o[7] = (_Float16)b.w;
    *(h8*)(dst + i) = o;
  }
}

// ---------------------------------------------------------------------------
// Stage 1: Q = (x Wq^T + bq)/8, K = x Wk^T + bk, V = x Wv^T + bv  (all f16 in).
// Q,K stored f16 [B,H,S,64]; V stored transposed f16 [B,H,64,S].
// One wave = one 16x16 output tile; A fragment (x) reused for 3 WMMAs.
// ---------------------------------------------------------------------------
__global__ __launch_bounds__(128)
void qkv_proj_wmma(const _Float16* __restrict__ xh,
                   const _Float16* __restrict__ Wqh, const float* __restrict__ bq,
                   const _Float16* __restrict__ Wkh, const float* __restrict__ bk,
                   const _Float16* __restrict__ Wvh, const float* __restrict__ bv,
                   _Float16* __restrict__ Qh, _Float16* __restrict__ Kh,
                   _Float16* __restrict__ Vt)
{
  const int lane = threadIdx.x & 31;
  const int wave = threadIdx.x >> 5;
  const int half = lane >> 4;
  const int lm   = lane & 15;
  const int m0 = blockIdx.x * 16;                  // row tile over B*S
  const int n0 = (blockIdx.y * 4 + wave) * 16;     // col tile over D

  const _Float16* xrow = xh  + (size_t)(m0 + lm) * D_;  // A: lane -> M=lm
  const _Float16* qrow = Wqh + (size_t)(n0 + lm) * D_;  // B: lane -> N=lm
  const _Float16* krow = Wkh + (size_t)(n0 + lm) * D_;
  const _Float16* vrow = Wvh + (size_t)(n0 + lm) * D_;

  v8f cq = {}, ck = {}, cv = {};
  const int kbA = half * 16;   // A: 16 consecutive K selected by lane half
  const int kbB = half * 16;   // B: 16 consecutive K selected by lane half

  for (int k = 0; k < D_; k += 32) {
    const v16h a   = *(const v16h*)(xrow + k + kbA);
    const v16h bqf = *(const v16h*)(qrow + k + kbB);
    const v16h bkf = *(const v16h*)(krow + k + kbB);
    const v16h bvf = *(const v16h*)(vrow + k + kbB);
    cq = __builtin_amdgcn_wmma_f32_16x16x32_f16(false, a, false, bqf, (short)0, cq, false, false);
    ck = __builtin_amdgcn_wmma_f32_16x16x32_f16(false, a, false, bkf, (short)0, ck, false, false);
    cv = __builtin_amdgcn_wmma_f32_16x16x32_f16(false, a, false, bvf, (short)0, cv, false, false);
  }

  // C layout: lane holds column N = lm, rows M = r + 8*half
  const int n = n0 + lm;
  const int h = n >> 6;
  const int d = n & 63;
  const float bqv = bq[n], bkv = bk[n], bvv = bv[n];

#pragma unroll
  for (int r = 0; r < 8; ++r) {
    const int m = m0 + r + half * 8;
    const int b = m >> 11;
    const int s = m & 2047;
    const size_t rowoff = ((size_t)(b * H_ + h) * S_ + s) * DH_ + d;
    Qh[rowoff] = (_Float16)((cq[r] + bqv) * 0.125f);   // fold in 1/sqrt(DH)
    Kh[rowoff] = (_Float16)(ck[r] + bkv);
    Vt[((size_t)(b * H_ + h) * DH_ + d) * S_ + s] = (_Float16)(cv[r] + bvv);
  }
}

// ---------------------------------------------------------------------------
// Stage 2: flash attention. Block = 4 waves x 16 queries; keys in blocks of 32.
// K/V tiles staged to LDS by the Tensor Data Mover (TENSORcnt-synced).
// ---------------------------------------------------------------------------
__global__ __launch_bounds__(128)
void flash_attn_wmma(const _Float16* __restrict__ Qh,
                     const _Float16* __restrict__ Kh,
                     const _Float16* __restrict__ Vt,
                     float* __restrict__ out)
{
  __shared__ _Float16 sK[32 * 64];     // [kk][d]   key tile
  __shared__ _Float16 sV[64 * 32];     // [d][kk]   value tile (transposed)
  __shared__ _Float16 sP[4][16 * 32];  // per-wave probs [q][kk]

  const int tid  = threadIdx.x;
  const int lane = tid & 31;
  const int wave = tid >> 5;
  const int half = lane >> 4;
  const int lm   = lane & 15;

  const int b  = blockIdx.z;
  const int h  = blockIdx.y;
  const int q0 = blockIdx.x * 64 + wave * 16;

  const size_t bh = (size_t)(b * H_ + h);
  const _Float16* Qbh = Qh + bh * (size_t)S_ * DH_;
  const _Float16* Kbh = Kh + bh * (size_t)S_ * DH_;
  const _Float16* Vbh = Vt + bh * (size_t)DH_ * S_;

#if USE_TDM
  const unsigned ldsK = (unsigned)(size_t)&sK[0];
  const unsigned ldsV = (unsigned)(size_t)&sV[0];
#endif

  // Q fragments (16 queries x 64 dims) held in registers for the whole loop.
  const _Float16* qp = Qbh + (size_t)(q0 + lm) * DH_;
  const int kbA = half * 16;
  const v16h qf0 = *(const v16h*)(qp + kbA);
  const v16h qf1 = *(const v16h*)(qp + 32 + kbA);

  v8f o0 = {}, o1 = {}, o2 = {}, o3 = {};
  float mrow[8], lrow[8];
#pragma unroll
  for (int r = 0; r < 8; ++r) { mrow[r] = -1e30f; lrow[r] = 0.f; }

  _Float16* pb = &sP[wave][0];
  const float LOG2E = 1.44269504088896340736f;

  for (int kk0 = 0; kk0 < S_; kk0 += 32) {
    __syncthreads();   // previous iteration's LDS reads complete before overwrite
#if USE_TDM
    if (wave == 0) {   // one wave drives the DMA engine (EXEC ignored by TDM)
      // K tile: 32 rows x 64 elems from row-major [S,64] tensor.
      tdm_load_2d(ldsK, Kbh + (size_t)kk0 * DH_, DH_, S_, DH_, 32, DH_);
      // V^T tile: 64 rows x 32 elems from row-major [64,S] tensor.
      tdm_load_2d(ldsV, Vbh + kk0, S_, DH_, 32, DH_, S_);
#  if __has_builtin(__builtin_amdgcn_s_wait_tensorcnt)
      __builtin_amdgcn_s_wait_tensorcnt(0);
#  else
      asm volatile("s_wait_tensorcnt 0" ::: "memory");
#  endif
    }
#else
    {
      const int row = tid >> 2, col = (tid & 3) * 16;
      *(h8*)&sK[row * 64 + col]     = *(const h8*)&Kbh[(size_t)(kk0 + row) * DH_ + col];
      *(h8*)&sK[row * 64 + col + 8] = *(const h8*)&Kbh[(size_t)(kk0 + row) * DH_ + col + 8];
    }
    {
      const int row = tid >> 1, col = (tid & 1) * 16;
      *(h8*)&sV[row * 32 + col]     = *(const h8*)&Vbh[(size_t)row * S_ + kk0 + col];
      *(h8*)&sV[row * 32 + col + 8] = *(const h8*)&Vbh[(size_t)row * S_ + kk0 + col + 8];
    }
#endif
    if (kk0 + 32 < S_) {  // hint next key tile toward the caches
      __builtin_prefetch(&Kbh[(size_t)(kk0 + 32) * DH_ + (tid * 16 & 2047)], 0, 3);
    }
    __syncthreads();

    // scores S[16q x 32k]: two 16x16 n-tiles, each 2 WMMAs over d=64.
    v8f s0 = {}, s1 = {};
    {
      const int doff = half * 16;
      const v16h b00 = *(const v16h*)&sK[lm * 64 + doff];
      const v16h b10 = *(const v16h*)&sK[(16 + lm) * 64 + doff];
      const v16h b01 = *(const v16h*)&sK[lm * 64 + 32 + doff];
      const v16h b11 = *(const v16h*)&sK[(16 + lm) * 64 + 32 + doff];
      s0 = __builtin_amdgcn_wmma_f32_16x16x32_f16(false, qf0, false, b00, (short)0, s0, false, false);
      s0 = __builtin_amdgcn_wmma_f32_16x16x32_f16(false, qf1, false, b01, (short)0, s0, false, false);
      s1 = __builtin_amdgcn_wmma_f32_16x16x32_f16(false, qf0, false, b10, (short)0, s1, false, false);
      s1 = __builtin_amdgcn_wmma_f32_16x16x32_f16(false, qf1, false, b11, (short)0, s1, false, false);
    }

    // Online softmax. C-fragment rows r+8*half live across a 16-lane group,
    // so xor-shuffles with masks 8..1 reduce within that group (wave32).
#pragma unroll
    for (int r = 0; r < 8; ++r) {
      float mx = fmaxf(s0[r], s1[r]);
#pragma unroll
      for (int msk = 8; msk >= 1; msk >>= 1) mx = fmaxf(mx, __shfl_xor(mx, msk, 32));
      const float mnew = fmaxf(mrow[r], mx);
      const float sf   = exp2f((mrow[r] - mnew) * LOG2E);
      mrow[r] = mnew;
      const float p0 = exp2f((s0[r] - mnew) * LOG2E);
      const float p1 = exp2f((s1[r] - mnew) * LOG2E);
      float rs = p0 + p1;
#pragma unroll
      for (int msk = 8; msk >= 1; msk >>= 1) rs += __shfl_xor(rs, msk, 32);
      lrow[r] = lrow[r] * sf + rs;
      pb[(r + 8 * half) * 32 + lm]      = (_Float16)p0;  // restripe C->A via LDS
      pb[(r + 8 * half) * 32 + 16 + lm] = (_Float16)p1;
      o0[r] *= sf; o1[r] *= sf; o2[r] *= sf; o3[r] *= sf;
    }

    // Same-wave LDS ops are in-order; make the compiler respect it too.
    asm volatile("s_wait_dscnt 0" ::: "memory");

    // P (A-fragment) and V^T (B-fragments): O += P x V, 4 WMMAs over d tiles.
    const v16h pa = *(const v16h*)&pb[lm * 32 + kbA];
    const int koff = half * 16;
    const v16h vb0 = *(const v16h*)&sV[(0 * 16 + lm) * 32 + koff];
    const v16h vb1 = *(const v16h*)&sV[(1 * 16 + lm) * 32 + koff];
    const v16h vb2 = *(const v16h*)&sV[(2 * 16 + lm) * 32 + koff];
    const v16h vb3 = *(const v16h*)&sV[(3 * 16 + lm) * 32 + koff];
    o0 = __builtin_amdgcn_wmma_f32_16x16x32_f16(false, pa, false, vb0, (short)0, o0, false, false);
    o1 = __builtin_amdgcn_wmma_f32_16x16x32_f16(false, pa, false, vb1, (short)0, o1, false, false);
    o2 = __builtin_amdgcn_wmma_f32_16x16x32_f16(false, pa, false, vb2, (short)0, o2, false, false);
    o3 = __builtin_amdgcn_wmma_f32_16x16x32_f16(false, pa, false, vb3, (short)0, o3, false, false);
  }

  // Normalize and write merged-head output [B,S,D] fp32.
#pragma unroll
  for (int r = 0; r < 8; ++r) {
    const float inv = 1.0f / lrow[r];
    const int q = q0 + r + 8 * half;
    float* op = out + ((size_t)b * S_ + q) * D_ + h * DH_;
    op[0 * 16 + lm] = o0[r] * inv;
    op[1 * 16 + lm] = o1[r] * inv;
    op[2 * 16 + lm] = o2[r] * inv;
    op[3 * 16 + lm] = o3[r] * inv;
  }
}

// ---------------------------------------------------------------------------
extern "C" void kernel_launch(void* const* d_in, const int* in_sizes, int n_in,
                              void* d_out, int out_size, void* d_ws, size_t ws_size,
                              hipStream_t stream)
{
  const float* x  = (const float*)d_in[0];
  const float* Wq = (const float*)d_in[1];
  const float* bq = (const float*)d_in[2];
  const float* Wk = (const float*)d_in[3];
  const float* bk = (const float*)d_in[4];
  const float* Wv = (const float*)d_in[5];
  const float* bv = (const float*)d_in[6];
  float* out = (float*)d_out;

  const size_t per  = (size_t)B_ * H_ * S_ * DH_;  // 6.29M f16 elements
  const size_t wsz  = (size_t)D_ * D_;             // 589K f16 elements
  _Float16* Qh  = (_Float16*)d_ws;
  _Float16* Kh  = Qh + per;
  _Float16* Vt  = Kh + per;
  _Float16* xh  = Vt + per;
  _Float16* Wqh = xh + per;
  _Float16* Wkh = Wqh + wsz;
  _Float16* Wvh = Wkh + wsz;

  const int nx = (int)per;   // B*S*D
  const int nw = (int)wsz;   // D*D
  cvt_f16<<<nx / (256 * 8), 256, 0, stream>>>(x,  xh,  nx);
  cvt_f16<<<nw / (256 * 8), 256, 0, stream>>>(Wq, Wqh, nw);
  cvt_f16<<<nw / (256 * 8), 256, 0, stream>>>(Wk, Wkh, nw);
  cvt_f16<<<nw / (256 * 8), 256, 0, stream>>>(Wv, Wvh, nw);

  dim3 g1((B_ * S_) / 16, D_ / 64);   // 512 x 12 blocks, 4 waves = 4 n-tiles each
  qkv_proj_wmma<<<g1, 128, 0, stream>>>(xh, Wqh, bq, Wkh, bk, Wvh, bv, Qh, Kh, Vt);

  dim3 g2(S_ / 64, H_, B_);           // 32 x 12 x 4 blocks of 4 waves
  flash_attn_wmma<<<g2, 128, 0, stream>>>(Qh, Kh, Vt, out);
}
